// MultiheadAttention_22436909154627
// MI455X (gfx1250) — compile-verified, matrix operand order checked by reading, not run on hardware
//
#include <hip/hip_runtime.h>

// ---------------------------------------------------------------------------
// MultiheadAttention forward on gfx1250 (MI455X), bf16 WMMA + f32 accumulate.
//   B=2, N=2048, EMBED=1024, HEADS=16, HEAD_DIM=64
// ---------------------------------------------------------------------------

#define EMBED 1024
#define HEADS 16
#define HDIM  64
#define SEQ   2048
#define BATCH 2
#define NTOK  (BATCH * SEQ)   // 4096 tokens
#define QKVD  (3 * EMBED)     // 3072
#define ATT_SCALE 0.125f      // 64^-0.5

typedef __attribute__((ext_vector_type(4)))  __bf16 v4bf;
typedef __attribute__((ext_vector_type(8)))  __bf16 v8bf;
typedef __attribute__((ext_vector_type(16))) __bf16 v16bf;
typedef __attribute__((ext_vector_type(8)))  float  v8f;

union BF16Frag { v16bf v; v8bf h[2]; };

// Load a 16x32 bf16 WMMA A/B fragment from a row-major [rows x stride] matrix.
// Per CDNA5 ISA layout: lanes 0-15 hold cols [c0..c0+7] & [c0+16..c0+23],
// lanes 16-31 hold cols [c0+8..c0+15] & [c0+24..c0+31]; row = caller-supplied.
__device__ __forceinline__ v16bf load_frag(const __bf16* __restrict__ base,
                                           int row, int col0, int stride, int lane) {
  const int hi = (lane >> 4) & 1;
  const __bf16* p = base + (size_t)row * stride + col0 + (hi << 3);
  BF16Frag f;
  f.h[0] = *reinterpret_cast<const v8bf*>(p);
  f.h[1] = *reinterpret_cast<const v8bf*>(p + 16);
  return f.v;
}

__device__ __forceinline__ v8f wmma_bf16(v16bf a, v16bf b, v8f c) {
  return __builtin_amdgcn_wmma_f32_16x16x32_bf16(false, a, false, b, (short)0, c,
                                                 false, false);
}

// ---------------------------------------------------------------------------
// Kernel 1: f32 -> bf16 cast (vectorized x4; sizes are multiples of 1024)
// ---------------------------------------------------------------------------
__global__ __launch_bounds__(256)
void cast_f32_bf16(const float* __restrict__ in, __bf16* __restrict__ out, int n) {
  int i = (blockIdx.x * blockDim.x + threadIdx.x) << 2;
  if (i >= n) return;
  float4 f = *reinterpret_cast<const float4*>(in + i);
  v4bf o;
  o[0] = (__bf16)f.x; o[1] = (__bf16)f.y; o[2] = (__bf16)f.z; o[3] = (__bf16)f.w;
  *reinterpret_cast<v4bf*>(out + i) = o;
}

// ---------------------------------------------------------------------------
// Kernel 2: QKV projection.  qkv[t,d] = x[t,:] . w_qkv[d,:] + b_qkv[d]
//   x: [4096,1024] bf16, w_qkv: [3072,1024] bf16.
//   Q,K -> [B,H,N,64] bf16 ; V -> transposed [B,H,64,N] bf16.
// Wave tile: 32 tokens x 32 dims (2x2 WMMA tiles). WG = 8 waves = 64 x 128.
// ---------------------------------------------------------------------------
__global__ __launch_bounds__(256)
void qkv_gemm(const __bf16* __restrict__ xb, const __bf16* __restrict__ wqb,
              const float* __restrict__ bias,
              __bf16* __restrict__ q_ws, __bf16* __restrict__ k_ws,
              __bf16* __restrict__ vt_ws) {
  const int lane = threadIdx.x & 31;
  const int wave = threadIdx.x >> 5;
  const int m0 = blockIdx.y * 64 + (wave & 1) * 32;   // token base
  const int n0 = blockIdx.x * 128 + (wave >> 1) * 32; // qkv-dim base
  const int lrow = lane & 15;

  v8f acc[2][2];
  acc[0][0] = acc[0][1] = acc[1][0] = acc[1][1] = v8f{};

  for (int k = 0; k < EMBED; k += 32) {
    v16bf a0 = load_frag(xb,  m0 + lrow,      k, EMBED, lane);
    v16bf a1 = load_frag(xb,  m0 + 16 + lrow, k, EMBED, lane);
    v16bf b0 = load_frag(wqb, n0 + lrow,      k, EMBED, lane);
    v16bf b1 = load_frag(wqb, n0 + 16 + lrow, k, EMBED, lane);
    acc[0][0] = wmma_bf16(a0, b0, acc[0][0]);
    acc[0][1] = wmma_bf16(a0, b1, acc[0][1]);
    acc[1][0] = wmma_bf16(a1, b0, acc[1][0]);
    acc[1][1] = wmma_bf16(a1, b1, acc[1][1]);
  }

#pragma unroll
  for (int mt = 0; mt < 2; ++mt) {
#pragma unroll
    for (int nt = 0; nt < 2; ++nt) {
      const int dcol  = n0 + nt * 16 + lrow;      // qkv output dim, lane = column
      const float bv  = bias[dcol];
      const int which = dcol >> 10;               // 0=Q 1=K 2=V
      const int h     = (dcol >> 6) & (HEADS - 1);
      const int dh    = dcol & (HDIM - 1);
      const int tokb  = m0 + mt * 16 + ((lane >> 4) << 3);
      const int b     = tokb >> 11;               // /SEQ
      const int nn    = tokb & (SEQ - 1);
      v8f c = acc[mt][nt];
      if (which == 2) {
        // V transposed: fixed dh, 8 consecutive tokens -> one 16B store
        v8bf o;
#pragma unroll
        for (int r = 0; r < 8; ++r) o[r] = (__bf16)(c[r] + bv);
        __bf16* dst = vt_ws + ((size_t)(b * HEADS + h) * HDIM + dh) * SEQ + nn;
        *reinterpret_cast<v8bf*>(dst) = o;
      } else {
        __bf16* dst = (which == 0 ? q_ws : k_ws);
#pragma unroll
        for (int r = 0; r < 8; ++r) {
          dst[((size_t)(b * HEADS + h) * SEQ + (nn + r)) * HDIM + dh] =
              (__bf16)(c[r] + bv);
        }
      }
    }
  }
}

// ---------------------------------------------------------------------------
// Kernel 3: flash attention (one wave = 16 queries of one (b,h); 8 waves/WG).
// Computes S^T = K.Q^T so each lane owns one query column: softmax reductions
// are 8-wide in-lane + one shfl_xor(16); P^T f32 accumulators repack in-lane
// into the exact B-fragment layout for the P.V WMMA (no cross-lane moves).
// ---------------------------------------------------------------------------
__global__ __launch_bounds__(256)
void flash_attn(const __bf16* __restrict__ q_ws, const __bf16* __restrict__ k_ws,
                const __bf16* __restrict__ vt_ws, __bf16* __restrict__ o_ws) {
  const int lane = threadIdx.x & 31;
  const int wave = threadIdx.x >> 5;
  const int bh   = blockIdx.y;                    // 0..B*H-1
  const int q0   = blockIdx.x * 128 + wave * 16;  // query base for this wave
  const int lrow = lane & 15;

  const __bf16* Q  = q_ws  + (size_t)bh * SEQ * HDIM;
  const __bf16* K  = k_ws  + (size_t)bh * SEQ * HDIM;
  const __bf16* VT = vt_ws + (size_t)bh * HDIM * SEQ;

  // Q fragments (B operand): held in registers for the whole key loop.
  const v16bf qf0 = load_frag(Q, q0 + lrow, 0,  HDIM, lane);
  const v16bf qf1 = load_frag(Q, q0 + lrow, 32, HDIM, lane);

  float m_run = -1e30f, l_run = 0.f;
  v8f o_acc[4];
  o_acc[0] = o_acc[1] = o_acc[2] = o_acc[3] = v8f{};

  for (int k0 = 0; k0 < SEQ; k0 += 32) {
    // Prefetch next chunk of K / V^T into the cache hierarchy.
    if (k0 + 32 < SEQ) {
      __builtin_prefetch(K + (size_t)(k0 + 32 + lrow) * HDIM, 0, 1);
      __builtin_prefetch(VT + (size_t)lrow * SEQ + (k0 + 32), 0, 1);
    }
    // S^T tiles: M = key, N = query.
    v16bf kf00 = load_frag(K, k0 + lrow,      0,  HDIM, lane);
    v16bf kf01 = load_frag(K, k0 + lrow,      32, HDIM, lane);
    v16bf kf10 = load_frag(K, k0 + 16 + lrow, 0,  HDIM, lane);
    v16bf kf11 = load_frag(K, k0 + 16 + lrow, 32, HDIM, lane);
    v8f s0 = v8f{};
    s0 = wmma_bf16(kf00, qf0, s0);
    s0 = wmma_bf16(kf01, qf1, s0);
    v8f s1 = v8f{};
    s1 = wmma_bf16(kf10, qf0, s1);
    s1 = wmma_bf16(kf11, qf1, s1);

    // Online softmax (per lane: 16 scores of one query; partner lane ^16 has
    // the other 16 key rows of the same query).
    float mloc = -1e30f;
#pragma unroll
    for (int r = 0; r < 8; ++r) {
      s0[r] *= ATT_SCALE;
      s1[r] *= ATT_SCALE;
      mloc = fmaxf(mloc, fmaxf(s0[r], s1[r]));
    }
    mloc = fmaxf(mloc, __shfl_xor(mloc, 16, 32));
    const float m_new = fmaxf(m_run, mloc);
    const float corr  = __expf(m_run - m_new);

    float psum = 0.f;
    v16bf pf;  // P^T B-fragment: elems 0..7 <- key tile 0, 8..15 <- key tile 1
#pragma unroll
    for (int r = 0; r < 8; ++r) {
      const float p0 = __expf(s0[r] - m_new);
      const float p1 = __expf(s1[r] - m_new);
      psum += p0 + p1;
      pf[r]     = (__bf16)p0;
      pf[r + 8] = (__bf16)p1;
    }
    psum += __shfl_xor(psum, 16, 32);
    l_run = l_run * corr + psum;
    m_run = m_new;

#pragma unroll
    for (int t = 0; t < 4; ++t) {
#pragma unroll
      for (int r = 0; r < 8; ++r) o_acc[t][r] *= corr;
    }
    // O^T += V^T . P^T  (A = rows of V^T, 4 tiles cover head dim 64)
#pragma unroll
    for (int t = 0; t < 4; ++t) {
      v16bf vf = load_frag(VT, t * 16 + lrow, k0, SEQ, lane);
      o_acc[t] = wmma_bf16(vf, pf, o_acc[t]);
    }
  }

  // Epilogue: O[q, d] = O^T / l ; 8 consecutive d per lane -> one 16B store.
  const float inv = 1.0f / l_run;
  const int h = bh & (HEADS - 1);
  const int b = bh >> 4;
  const int q = q0 + lrow;
#pragma unroll
  for (int t = 0; t < 4; ++t) {
    v8bf o;
#pragma unroll
    for (int r = 0; r < 8; ++r) o[r] = (__bf16)(o_acc[t][r] * inv);
    const int d = h * HDIM + t * 16 + ((lane >> 4) << 3);
    *reinterpret_cast<v8bf*>(o_ws + (size_t)(b * SEQ + q) * EMBED + d) = o;
  }
}

// ---------------------------------------------------------------------------
// Kernel 4: output projection.  out[t,c] = o[t,:] . w_proj[c,:] + b_proj[c]
// f32 output straight to d_out. Same wave tiling as the QKV GEMM.
// ---------------------------------------------------------------------------
__global__ __launch_bounds__(256)
void proj_gemm(const __bf16* __restrict__ ob, const __bf16* __restrict__ wpb,
               const float* __restrict__ bias, float* __restrict__ out) {
  const int lane = threadIdx.x & 31;
  const int wave = threadIdx.x >> 5;
  const int m0 = blockIdx.y * 64 + (wave & 1) * 32;
  const int n0 = blockIdx.x * 128 + (wave >> 1) * 32;
  const int lrow = lane & 15;

  v8f acc[2][2];
  acc[0][0] = acc[0][1] = acc[1][0] = acc[1][1] = v8f{};

  for (int k = 0; k < EMBED; k += 32) {
    v16bf a0 = load_frag(ob,  m0 + lrow,      k, EMBED, lane);
    v16bf a1 = load_frag(ob,  m0 + 16 + lrow, k, EMBED, lane);
    v16bf b0 = load_frag(wpb, n0 + lrow,      k, EMBED, lane);
    v16bf b1 = load_frag(wpb, n0 + 16 + lrow, k, EMBED, lane);
    acc[0][0] = wmma_bf16(a0, b0, acc[0][0]);
    acc[0][1] = wmma_bf16(a0, b1, acc[0][1]);
    acc[1][0] = wmma_bf16(a1, b0, acc[1][0]);
    acc[1][1] = wmma_bf16(a1, b1, acc[1][1]);
  }

#pragma unroll
  for (int mt = 0; mt < 2; ++mt) {
#pragma unroll
    for (int nt = 0; nt < 2; ++nt) {
      const int col  = n0 + nt * 16 + lrow;
      const float bv = bias[col];
      const int tokb = m0 + mt * 16 + ((lane >> 4) << 3);
#pragma unroll
      for (int r = 0; r < 8; ++r)
        out[(size_t)(tokb + r) * EMBED + col] = acc[mt][nt][r] + bv;
    }
  }
}

// ---------------------------------------------------------------------------
extern "C" void kernel_launch(void* const* d_in, const int* in_sizes, int n_in,
                              void* d_out, int out_size, void* d_ws, size_t ws_size,
                              hipStream_t stream) {
  const float* x      = (const float*)d_in[0];  // [2,2048,1024]
  const float* w_qkv  = (const float*)d_in[1];  // [3072,1024]
  const float* b_qkv  = (const float*)d_in[2];  // [3072]
  const float* w_proj = (const float*)d_in[3];  // [1024,1024]
  const float* b_proj = (const float*)d_in[4];  // [1024]
  float* out = (float*)d_out;                   // [2,2048,1024]

  // Workspace partition (all sizes 16B-aligned). Total = 48 MB.
  char* ws = (char*)d_ws;
  __bf16* xb  = (__bf16*)ws; ws += (size_t)NTOK * EMBED * 2;   //  8 MB
  __bf16* wqb = (__bf16*)ws; ws += (size_t)QKVD * EMBED * 2;   //  6 MB
  __bf16* wpb = (__bf16*)ws; ws += (size_t)EMBED * EMBED * 2;  //  2 MB
  __bf16* q_ws  = (__bf16*)ws; ws += (size_t)NTOK * HDIM * HEADS * 2; // 8 MB
  __bf16* k_ws  = (__bf16*)ws; ws += (size_t)NTOK * HDIM * HEADS * 2; // 8 MB
  __bf16* vt_ws = (__bf16*)ws; ws += (size_t)NTOK * HDIM * HEADS * 2; // 8 MB
  __bf16* o_ws  = (__bf16*)ws;                                        // 8 MB

  // 1) casts to bf16 (4 elems/thread)
  {
    int n = NTOK * EMBED;
    cast_f32_bf16<<<n / 1024, 256, 0, stream>>>(x, xb, n);
    n = QKVD * EMBED;
    cast_f32_bf16<<<n / 1024, 256, 0, stream>>>(w_qkv, wqb, n);
    n = EMBED * EMBED;
    cast_f32_bf16<<<n / 1024, 256, 0, stream>>>(w_proj, wpb, n);
  }
  // 2) QKV projection: grid 24 x 64 WGs of 256
  qkv_gemm<<<dim3(QKVD / 128, NTOK / 64), 256, 0, stream>>>(
      xb, wqb, b_qkv, q_ws, k_ws, vt_ws);
  // 3) flash attention: grid 16 x (B*H=32)
  flash_attn<<<dim3(SEQ / 128, BATCH * HEADS), 256, 0, stream>>>(
      q_ws, k_ws, vt_ws, o_ws);
  // 4) output projection: grid 8 x 64
  proj_gemm<<<dim3(EMBED / 128, NTOK / 64), 256, 0, stream>>>(
      o_ws, wpb, b_proj, out);
}
